// MultiFeaturesMemoryBlock_17557826306474
// MI455X (gfx1250) — matrix operand order repaired and checked by reading, not compiled, not run on hardware
//
#include <hip/hip_runtime.h>
#include <math.h>

typedef __attribute__((ext_vector_type(16))) _Float16 v16h;
typedef __attribute__((ext_vector_type(8)))  _Float16 v8h;
typedef __attribute__((ext_vector_type(8)))  float    v8f;

#define BDIM   2
#define LSEQ   2048
#define MMEM   256
#define SSEQ   2304      // M + L
#define HDIM   512
#define DIEXP  1024      // EXP * H
#define DSTATE 16
#define DCONV  4

__device__ __forceinline__ float gelu_exact(float v) {
  return 0.5f * v * (1.0f + erff(v * 0.70710678118654752f));
}
__device__ __forceinline__ float silu_f(float v) {
  return v / (1.0f + __expf(-v));
}

#define EPI_NONE      0
#define EPI_BIAS      1
#define EPI_BIAS_GELU 2
#define EPI_FINAL     3

// C(128x64 tile, f32) = A(MxK, f16 row-major) * BT(NxK, f16 row-major == B^T).
// Tiles staged global->LDS with global_load_async_to_lds_b128, double buffered
// (ASYNCcnt), consumed by v_wmma_f32_16x16x32_f16. 8 waves = 4(M) x 2(N),
// each wave 32x32 (2x2 WMMA with A/B fragment reuse).
template<int EPI, bool FLIPA, bool FLIPOUT>
__global__ __launch_bounds__(256)
void gemm_wmma(const _Float16* __restrict__ A, const _Float16* __restrict__ BT,
               const float* __restrict__ bias, float* __restrict__ C,
               int N, int K, int Sflip,
               const float* __restrict__ xres, float* __restrict__ out2)
{
  __shared__ __align__(16) _Float16 Ah[2][128 * 40];
  __shared__ __align__(16) _Float16 Bh[2][64 * 40];
  const int tid  = threadIdx.x;
  const int lane = tid & 31;
  const int wv   = tid >> 5;
  const int lm   = lane & 15;
  const int lg   = lane >> 4;
  const int mw   = wv & 3;    // wave position along M (4 x 32 rows)
  const int nw   = wv >> 2;   // wave position along N (2 x 32 cols)
  const int rowBase = blockIdx.x * 128;
  const int colBase = blockIdx.y * 64;

  // Staging tasks (per 32-deep K tile):
  //   A: 128 rows x 4 chunks(8 halves) = 512 tasks -> p = tid, tid+256
  //   B: 64 rows x 4 chunks            = 256 tasks -> p = tid
  int ra0 = rowBase + (tid >> 2);
  int ra1 = rowBase + ((tid + 256) >> 2);
  if (FLIPA) {
    int b0 = ra0 / Sflip, s0 = ra0 - b0 * Sflip;
    ra0 = b0 * Sflip + (Sflip - 1 - s0);
    int b1 = ra1 / Sflip, s1 = ra1 - b1 * Sflip;
    ra1 = b1 * Sflip + (Sflip - 1 - s1);
  }
  const int q8 = (tid & 3) * 8;
  const _Float16* ga0 = A + (size_t)ra0 * K + q8;
  const _Float16* ga1 = A + (size_t)ra1 * K + q8;
  const _Float16* gb0 = BT + (size_t)(colBase + (tid >> 2)) * K + q8;
  uint32_t la0[2], la1[2], lb0[2];
  la0[0] = (uint32_t)(uintptr_t)&Ah[0][(tid >> 2) * 40 + q8];
  la0[1] = (uint32_t)(uintptr_t)&Ah[1][(tid >> 2) * 40 + q8];
  la1[0] = (uint32_t)(uintptr_t)&Ah[0][((tid + 256) >> 2) * 40 + q8];
  la1[1] = (uint32_t)(uintptr_t)&Ah[1][((tid + 256) >> 2) * 40 + q8];
  lb0[0] = (uint32_t)(uintptr_t)&Bh[0][(tid >> 2) * 40 + q8];
  lb0[1] = (uint32_t)(uintptr_t)&Bh[1][(tid >> 2) * 40 + q8];

  const int T = K >> 5;
  v8f acc[2][2] = {};

  auto stage = [&](int t, int buf) {
    const uint64_t a0 = (uint64_t)(uintptr_t)(ga0 + t * 32);
    const uint64_t a1 = (uint64_t)(uintptr_t)(ga1 + t * 32);
    const uint64_t b0 = (uint64_t)(uintptr_t)(gb0 + t * 32);
    asm volatile("global_load_async_to_lds_b128 %0, %1, off"
                 :: "v"(la0[buf]), "v"(a0) : "memory");
    asm volatile("global_load_async_to_lds_b128 %0, %1, off"
                 :: "v"(la1[buf]), "v"(a1) : "memory");
    asm volatile("global_load_async_to_lds_b128 %0, %1, off"
                 :: "v"(lb0[buf]), "v"(b0) : "memory");
  };

  stage(0, 0);
  for (int t = 0; t < T; ++t) {
    const int buf = t & 1;
    if (t + 1 < T) {
      stage(t + 1, buf ^ 1);
      asm volatile("s_wait_asynccnt 0x3" ::: "memory");  // tile t landed
    } else {
      asm volatile("s_wait_asynccnt 0x0" ::: "memory");
    }
    __syncthreads();

    v16h a[2], b[2];
#pragma unroll
    for (int im = 0; im < 2; ++im) {
      const int r = mw * 32 + im * 16 + lm;
      v8h lo = *reinterpret_cast<const v8h*>(&Ah[buf][r * 40 + 8 * lg]);
      v8h hi = *reinterpret_cast<const v8h*>(&Ah[buf][r * 40 + 16 + 8 * lg]);
#pragma unroll
      for (int i = 0; i < 8; ++i) { a[im][i] = lo[i]; a[im][8 + i] = hi[i]; }
    }
#pragma unroll
    for (int jn = 0; jn < 2; ++jn) {
      const int n = nw * 32 + jn * 16 + lm;
      v8h lo = *reinterpret_cast<const v8h*>(&Bh[buf][n * 40 + 8 * lg]);
      v8h hi = *reinterpret_cast<const v8h*>(&Bh[buf][n * 40 + 16 + 8 * lg]);
#pragma unroll
      for (int i = 0; i < 8; ++i) { b[jn][i] = lo[i]; b[jn][8 + i] = hi[i]; }
    }
#pragma unroll
    for (int im = 0; im < 2; ++im)
#pragma unroll
      for (int jn = 0; jn < 2; ++jn)
        acc[im][jn] = __builtin_amdgcn_wmma_f32_16x16x32_f16(
            false, a[im], false, b[jn], (short)0, acc[im][jn], false, false);
    __syncthreads();
  }

#pragma unroll
  for (int im = 0; im < 2; ++im) {
#pragma unroll
    for (int jn = 0; jn < 2; ++jn) {
      const int col = colBase + nw * 32 + jn * 16 + lm;
#pragma unroll
      for (int r = 0; r < 8; ++r) {
        int row = rowBase + mw * 32 + im * 16 + lg * 8 + r;
        float v = acc[im][jn][r];
        if (EPI == EPI_BIAS || EPI == EPI_BIAS_GELU || EPI == EPI_FINAL) v += bias[col];
        if (EPI == EPI_BIAS_GELU) v = gelu_exact(v);
        int orow = row;
        if (FLIPOUT) {
          int bb = row / Sflip;
          int ss = row - bb * Sflip;
          orow = bb * Sflip + (Sflip - 1 - ss);
        }
        if (EPI == EPI_FINAL) {
          if (col < 512)
            C[(size_t)orow * 512 + col] =
                (xres[(size_t)orow * 512 + col] + v) * 0.70710678118654752f;
          else
            out2[(size_t)orow * 512 + (col - 512)] = v;
        } else {
          C[(size_t)orow * N + col] = v;
        }
      }
    }
  }
}

// f32 -> f16 copy
__global__ void cvt_f16(const float* __restrict__ in, _Float16* __restrict__ out, int n)
{
  int i = blockIdx.x * 256 + threadIdx.x;
  if (i < n) out[i] = (_Float16)in[i];
}

// weight (KxN, f32) -> (NxK, f16) transpose
__global__ void cvt_f16_T(const float* __restrict__ in, _Float16* __restrict__ out,
                          int K, int N)
{
  int i = blockIdx.x * 256 + threadIdx.x;
  if (i >= K * N) return;
  int n = i / K, k = i - n * K;
  out[i] = (_Float16)in[(size_t)k * N + n];
}

// hs(f32) = concat(mem, h); xm(f16) = hs*m + (m*wm + bm)*(1-m)
__global__ void build_xm(const float* __restrict__ memtmp, const float* __restrict__ h,
                         const float* __restrict__ mask, const float* __restrict__ wm,
                         const float* __restrict__ bm, float* __restrict__ hs,
                         _Float16* __restrict__ xm)
{
  int idx = blockIdx.x * 256 + threadIdx.x;
  if (idx >= BDIM * SSEQ * HDIM) return;
  int c = idx % HDIM;
  int rs = idx / HDIM;
  int s = rs % SSEQ;
  int b = rs / SSEQ;
  float v = (s < MMEM) ? memtmp[((size_t)(b * MMEM + s)) * HDIM + c]
                       : h[((size_t)(b * LSEQ + (s - MMEM))) * HDIM + c];
  float mk = mask[b * SSEQ + s];
  hs[idx] = v;
  xm[idx] = (_Float16)(v * mk + (mk * wm[c] + bm[c]) * (1.0f - mk));
}

// xs(f16) = silu(depthwise_conv4(xz[:, :, :DI]))
__global__ void conv_silu(const float* __restrict__ xz, const float* __restrict__ cw,
                          const float* __restrict__ cb, _Float16* __restrict__ xs)
{
  int idx = blockIdx.x * 256 + threadIdx.x;
  if (idx >= BDIM * SSEQ * DIEXP) return;
  int di = idx % DIEXP;
  int rs = idx / DIEXP;
  int s = rs % SSEQ;
  int b = rs / SSEQ;
  float a = cb[di];
#pragma unroll
  for (int k = 0; k < DCONV; ++k) {
    int sp = s - (DCONV - 1) + k;
    if (sp >= 0)
      a += xz[((size_t)(b * SSEQ + sp)) * (2 * DIEXP) + di] * cw[di * DCONV + k];
  }
  xs[idx] = (_Float16)silu_f(a);
}

// Selective scan: one lane per (b, di, ds); 16-lane shfl butterflies for the
// rank-32 dt projection and the h.C contraction. yv written in place over xs.
__global__ __launch_bounds__(256)
void scan_kernel(_Float16* __restrict__ xs, const float* __restrict__ dbl,
                 const float* __restrict__ xz, const float* __restrict__ dt_w,
                 const float* __restrict__ dt_b, const float* __restrict__ A_log,
                 const float* __restrict__ Dp)
{
  int t  = blockIdx.x * 256 + threadIdx.x;
  int ds = t & 15;
  int c  = t >> 4;
  int di = c % DIEXP;
  int b  = c / DIEXP;
  float Ac  = -__expf(A_log[di * DSTATE + ds]);
  float w0  = dt_w[ds * DIEXP + di];
  float w1  = dt_w[(ds + 16) * DIEXP + di];
  float dtb = dt_b[di];
  float Dv  = Dp[di];
  float h = 0.0f;
  for (int s = 0; s < SSEQ; ++s) {
    size_t rowd = ((size_t)(b * SSEQ + s)) * 64;
    float p = dbl[rowd + ds] * w0 + dbl[rowd + 16 + ds] * w1;
#pragma unroll
    for (int m = 8; m >= 1; m >>= 1) p += __shfl_xor(p, m, 32);
    float dtpre = p + dtb;
    float dt = (dtpre > 20.0f) ? dtpre : log1pf(__expf(dtpre));
    size_t xi = ((size_t)(b * SSEQ + s)) * DIEXP + di;
    float x = (float)xs[xi];
    float Btv = dbl[rowd + 32 + ds];
    float Ctv = dbl[rowd + 48 + ds];
    h = h * __expf(dt * Ac) + dt * x * Btv;
    float y = h * Ctv;
#pragma unroll
    for (int m = 8; m >= 1; m >>= 1) y += __shfl_xor(y, m, 32);
    if (ds == 0) {
      float zv = xz[((size_t)(b * SSEQ + s)) * (2 * DIEXP) + DIEXP + di];
      xs[xi] = (_Float16)((y + x * Dv) * silu_f(zv));
    }
  }
}

// comb (on the fly) -> LayerNorm + h -> g(f16) = gelu(sigmoid(gate)*tanh(filt))
__global__ __launch_bounds__(256)
void ln_gate(const float* __restrict__ d0, const float* __restrict__ d1,
             const float* __restrict__ hs, const float* __restrict__ mask,
             const float* __restrict__ h, const float* __restrict__ gamma,
             const float* __restrict__ beta, _Float16* __restrict__ g)
{
  __shared__ float red[256];
  int r = blockIdx.x;          // 0..B*L-1
  int tid = threadIdx.x;
  int b = r / LSEQ;
  int l = r - b * LSEQ;
  size_t row = (size_t)(b * SSEQ + MMEM + l);
  float mk = mask[b * SSEQ + MMEM + l];
  size_t i0 = row * HDIM + tid, i1 = i0 + 256;
  float v0 = (d0[i0] + d1[i0]) * mk + hs[i0] * (1.0f - mk);
  float v1 = (d0[i1] + d1[i1]) * mk + hs[i1] * (1.0f - mk);

  red[tid] = v0 + v1;
  __syncthreads();
  for (int o = 128; o > 0; o >>= 1) { if (tid < o) red[tid] += red[tid + o]; __syncthreads(); }
  float mean = red[0] * (1.0f / 512.0f);
  __syncthreads();
  red[tid] = v0 * v0 + v1 * v1;
  __syncthreads();
  for (int o = 128; o > 0; o >>= 1) { if (tid < o) red[tid] += red[tid + o]; __syncthreads(); }
  float var = red[0] * (1.0f / 512.0f) - mean * mean;
  float rstd = rsqrtf(var + 1e-5f);

  const float* hrow = h + (size_t)r * HDIM;
  float f0 = (v0 - mean) * rstd * gamma[tid] + beta[tid] + hrow[tid];
  float f1 = (v1 - mean) * rstd * gamma[tid + 256] + beta[tid + 256] + hrow[tid + 256];
  float o = (1.0f / (1.0f + __expf(-f0))) * tanhf(f1);
  g[(size_t)r * 256 + tid] = (_Float16)gelu_exact(o);
}

extern "C" void kernel_launch(void* const* d_in, const int* in_sizes, int n_in,
                              void* d_out, int out_size, void* d_ws, size_t ws_size,
                              hipStream_t stream)
{
  (void)in_sizes; (void)n_in; (void)out_size; (void)ws_size;
  const float* x    = (const float*)d_in[0];
  const float* mme  = (const float*)d_in[1];
  const float* mask = (const float*)d_in[2];
  const float* w1   = (const float*)d_in[3];
  const float* b1   = (const float*)d_in[4];
  const float* wp   = (const float*)d_in[5];
  const float* bp   = (const float*)d_in[6];
  const float* wm   = (const float*)d_in[7];
  const float* bm   = (const float*)d_in[8];
  const float* P[2][9];   // 0 in_w,1 conv_w,2 conv_b,3 xproj_w,4 dt_w,5 dt_b,6 A_log,7 D,8 out_w
  for (int i = 0; i < 9; ++i) { P[0][i] = (const float*)d_in[9 + i];
                                P[1][i] = (const float*)d_in[18 + i]; }
  const float* gamma = (const float*)d_in[27];
  const float* beta  = (const float*)d_in[28];
  const float* w2    = (const float*)d_in[29];
  const float* b2    = (const float*)d_in[30];

  // ---- workspace layout: f32 region then f16 region ----
  float* W = (float*)d_ws;
  size_t o = 0;
  float* hbuf = W + o;  o += (size_t)BDIM * LSEQ * HDIM;
  float* hsb  = W + o;  o += (size_t)BDIM * SSEQ * HDIM;
  float* xzd[2]; xzd[0] = W + o; o += (size_t)BDIM * SSEQ * 2 * DIEXP;
                 xzd[1] = W + o; o += (size_t)BDIM * SSEQ * 2 * DIEXP;
  float* dbd[2]; dbd[0] = W + o; o += (size_t)BDIM * SSEQ * 64;
                 dbd[1] = W + o; o += (size_t)BDIM * SSEQ * 64;
  float* dird[2]; dird[0] = W + o; o += (size_t)BDIM * SSEQ * HDIM;
                  dird[1] = W + o; o += (size_t)BDIM * SSEQ * HDIM;
  float* memtmp = dird[0];  // consumed before dird[0] is produced

  _Float16* HB = (_Float16*)(W + o);
  size_t oh = 0;
  _Float16* xh   = HB + oh; oh += (size_t)BDIM * LSEQ * HDIM;
  _Float16* mmeh = HB + oh; oh += (size_t)BDIM * MMEM * 256;
  _Float16* xmh  = HB + oh; oh += (size_t)BDIM * SSEQ * HDIM;
  _Float16* xsh[2]; xsh[0] = HB + oh; oh += (size_t)BDIM * SSEQ * DIEXP;
                    xsh[1] = HB + oh; oh += (size_t)BDIM * SSEQ * DIEXP;
  _Float16* gh   = HB + oh; oh += (size_t)BDIM * LSEQ * 256;
  _Float16* w1T  = HB + oh; oh += 512 * 512;
  _Float16* wpT  = HB + oh; oh += 512 * 256;
  _Float16* inwT[2]; inwT[0] = HB + oh; oh += 2048 * 512;
                     inwT[1] = HB + oh; oh += 2048 * 512;
  _Float16* xpT[2];  xpT[0]  = HB + oh; oh += 64 * 1024;
                     xpT[1]  = HB + oh; oh += 64 * 1024;
  _Float16* owT[2];  owT[0]  = HB + oh; oh += 512 * 1024;
                     owT[1]  = HB + oh; oh += 512 * 1024;
  _Float16* w2T  = HB + oh; oh += 1024 * 256;

  float* out0 = (float*)d_out;
  float* out1 = out0 + (size_t)BDIM * LSEQ * 512;

  dim3 blk(256);
  #define GRID1(n) dim3(((n) + 255) / 256)

  // ---- one-time f16 conversions / weight transposes ----
  cvt_f16<<<GRID1(BDIM*LSEQ*HDIM), blk, 0, stream>>>(x, xh, BDIM*LSEQ*HDIM);
  cvt_f16<<<GRID1(BDIM*MMEM*256), blk, 0, stream>>>(mme, mmeh, BDIM*MMEM*256);
  cvt_f16_T<<<GRID1(512*512),  blk, 0, stream>>>(w1, w1T, 512, 512);
  cvt_f16_T<<<GRID1(256*512),  blk, 0, stream>>>(wp, wpT, 256, 512);
  cvt_f16_T<<<GRID1(512*2048), blk, 0, stream>>>(P[0][0], inwT[0], 512, 2048);
  cvt_f16_T<<<GRID1(512*2048), blk, 0, stream>>>(P[1][0], inwT[1], 512, 2048);
  cvt_f16_T<<<GRID1(1024*64),  blk, 0, stream>>>(P[0][3], xpT[0], 1024, 64);
  cvt_f16_T<<<GRID1(1024*64),  blk, 0, stream>>>(P[1][3], xpT[1], 1024, 64);
  cvt_f16_T<<<GRID1(1024*512), blk, 0, stream>>>(P[0][8], owT[0], 1024, 512);
  cvt_f16_T<<<GRID1(1024*512), blk, 0, stream>>>(P[1][8], owT[1], 1024, 512);
  cvt_f16_T<<<GRID1(256*1024), blk, 0, stream>>>(w2, w2T, 256, 1024);

  // h = gelu(x @ w1 + b1)                     [4096 x 512 x 512]
  gemm_wmma<EPI_BIAS_GELU, false, false><<<dim3(32, 8), blk, 0, stream>>>(
      xh, w1T, b1, hbuf, 512, 512, SSEQ, nullptr, nullptr);
  // mem = mme @ wp + bp                       [512 x 256 x 512]
  gemm_wmma<EPI_BIAS, false, false><<<dim3(4, 8), blk, 0, stream>>>(
      mmeh, wpT, bp, memtmp, 512, 256, SSEQ, nullptr, nullptr);
  build_xm<<<GRID1(BDIM*SSEQ*HDIM), blk, 0, stream>>>(
      memtmp, hbuf, mask, wm, bm, hsb, xmh);

  for (int d = 0; d < 2; ++d) {
    // xz = xm(flipped for bwd) @ in_w         [4608 x 512 x 2048]
    if (d == 0)
      gemm_wmma<EPI_NONE, false, false><<<dim3(36, 32), blk, 0, stream>>>(
          xmh, inwT[0], nullptr, xzd[0], 2048, 512, SSEQ, nullptr, nullptr);
    else
      gemm_wmma<EPI_NONE, true, false><<<dim3(36, 32), blk, 0, stream>>>(
          xmh, inwT[1], nullptr, xzd[1], 2048, 512, SSEQ, nullptr, nullptr);
    conv_silu<<<GRID1(BDIM*SSEQ*DIEXP), blk, 0, stream>>>(
        xzd[d], P[d][1], P[d][2], xsh[d]);
    // dbl = xs @ xproj_w                      [4608 x 1024 x 64]
    gemm_wmma<EPI_NONE, false, false><<<dim3(36, 1), blk, 0, stream>>>(
        xsh[d], xpT[d], nullptr, dbd[d], 64, 1024, SSEQ, nullptr, nullptr);
    scan_kernel<<<128, blk, 0, stream>>>(
        xsh[d], dbd[d], xzd[d], P[d][4], P[d][5], P[d][6], P[d][7]);
    // dir_out = yv @ out_w                    [4608 x 1024 x 512]
    if (d == 0)
      gemm_wmma<EPI_NONE, false, false><<<dim3(36, 8), blk, 0, stream>>>(
          xsh[0], owT[0], nullptr, dird[0], 512, 1024, SSEQ, nullptr, nullptr);
    else
      gemm_wmma<EPI_NONE, false, true><<<dim3(36, 8), blk, 0, stream>>>(
          xsh[1], owT[1], nullptr, dird[1], 512, 1024, SSEQ, nullptr, nullptr);
  }

  // combine + LN + residual + gated activation -> g (f16)
  ln_gate<<<BDIM * LSEQ, blk, 0, stream>>>(
      dird[0], dird[1], hsb, mask, hbuf, gamma, beta, gh);
  // o = gelu(g) @ w2 + b2 -> ((x+res)/sqrt2, skip)   [4096 x 256 x 1024]
  gemm_wmma<EPI_FINAL, false, false><<<dim3(32, 16), blk, 0, stream>>>(
      gh, w2T, b2, out0, 1024, 256, SSEQ, x, out1);
}
